// FlashMHA_78486232367294
// MI455X (gfx1250) — matrix operand order repaired and checked by reading, not compiled, not run on hardware
//
#include <hip/hip_runtime.h>
#include <hip/hip_bf16.h>
#include <stdint.h>

// ---------------------------------------------------------------------------
// Fused MHA for MI455X (gfx1250, wave32, WMMA 16x16x32 bf16 / f32 accum)
// B=8, S=1024, E=1024, H=16, D=64
// Round 3: fix async-builtin signature: (as1 v4i*, as3 v4i*, imm, imm).
// ---------------------------------------------------------------------------

typedef __bf16 bf16_t;
typedef __attribute__((ext_vector_type(16))) __bf16 v16bf;
typedef __attribute__((ext_vector_type(8)))  float  v8f;

union FragBF {
  v16bf v;
  uint4 q[2];
};

constexpr int B_ = 8, S_ = 1024, E_ = 1024, H_ = 16, D_ = 64;

__device__ __forceinline__ unsigned short f2bf_bits(float f) {
  unsigned u = __builtin_bit_cast(unsigned, f);
  return (unsigned short)((u + 0x7FFFu + ((u >> 16) & 1u)) >> 16);  // RNE
}
__device__ __forceinline__ bf16_t f2bf(float f) {
  unsigned short b = f2bf_bits(f);
  return __builtin_bit_cast(bf16_t, b);
}

#if __has_builtin(__builtin_amdgcn_cvt_pk_bf16_f32)
typedef __attribute__((ext_vector_type(2))) __bf16 v2bf_t;
__device__ __forceinline__ unsigned pk2bf(float a, float b) {
  v2bf_t r = __builtin_amdgcn_cvt_pk_bf16_f32(a, b);
  return __builtin_bit_cast(unsigned, r);
}
#else
__device__ __forceinline__ unsigned pk2bf(float a, float b) {
  return (unsigned)f2bf_bits(a) | ((unsigned)f2bf_bits(b) << 16);
}
#endif

// ---- async global -> LDS (gfx1250 GLOBAL_LOAD_ASYNC_TO_LDS_B128) ----------
#if __has_builtin(__builtin_amdgcn_global_load_async_to_lds_b128)
#define HAVE_ASYNC_LDS 1
typedef __attribute__((ext_vector_type(4))) int v4i_t;
typedef __attribute__((address_space(1))) v4i_t as1_v4i;  // global int4
typedef __attribute__((address_space(3))) v4i_t as3_v4i;  // LDS int4
#else
#define HAVE_ASYNC_LDS 0
#endif

__device__ __forceinline__ void async_copy16(const bf16_t* g, bf16_t* l) {
#if HAVE_ASYNC_LDS
  __builtin_amdgcn_global_load_async_to_lds_b128((as1_v4i*)g, (as3_v4i*)l, 0, 0);
#else
  *(uint4*)l = *(const uint4*)g;
#endif
}

__device__ __forceinline__ void wait_async_lds() {
#if HAVE_ASYNC_LDS
#if __has_builtin(__builtin_amdgcn_s_wait_asynccnt)
  __builtin_amdgcn_s_wait_asynccnt(0);
#else
  asm volatile("s_wait_asynccnt 0" ::: "memory");
#endif
#endif
}

// ---------------------------------------------------------------------------
// Kernel 1: packed in-projection.  C[m, e_g] = X[m,:] . W_in[e_g,:] + bias
// X is q/k/v chosen by e_g/1024.  Q,K outputs -> bf16 [B,H,S,D];
// V output -> bf16 [B,H,D,S] (pre-transposed for attention's B-fragments).
// Block tile 128x128, BK=32, 8 waves (4M x 2N), wave tile 32x64.
// ---------------------------------------------------------------------------
__global__ __launch_bounds__(256) void mha_inproj(
    const float* __restrict__ q, const float* __restrict__ k,
    const float* __restrict__ v, const float* __restrict__ w_in,
    const float* __restrict__ b_in,
    bf16_t* __restrict__ qh, bf16_t* __restrict__ kh, bf16_t* __restrict__ vt)
{
  __shared__ __align__(16) bf16_t Asm[128 * 40];  // rows padded to 80B
  __shared__ __align__(16) bf16_t Bsm[128 * 40];

  const int m0  = blockIdx.x * 128;        // row among B*S = 8192
  const int n0g = blockIdx.y * 128;        // col among 3E = 3072
  const int t   = n0g >> 10;               // 0:q 1:k 2:v
  const float* __restrict__ A = (t == 0) ? q : (t == 1) ? k : v;
  bf16_t* __restrict__ dst    = (t == 0) ? qh : (t == 1) ? kh : vt;
  const float* __restrict__ W = w_in + (size_t)n0g * E_;

  const int tid  = threadIdx.x;
  const int w    = tid >> 5;
  const int lane = tid & 31;
  const int ln   = lane & 15;
  const int half = lane >> 4;
  const int wm   = w >> 1;                 // 0..3
  const int wn   = w & 1;                  // 0..1

  v8f acc[2][4] = {};

  for (int kk = 0; kk < E_; kk += 32) {
    __syncthreads();
    #pragma unroll
    for (int i = 0; i < 4; ++i) {
      int li  = i * 256 + tid;             // 1024 float4 slots per tile
      int row = li >> 3;                   // 0..127
      int c4  = (li & 7) * 4;              // 0..28
      float4 av = *(const float4*)(A + (size_t)(m0 + row) * E_ + kk + c4);
      *(uint2*)(&Asm[row * 40 + c4]) = make_uint2(pk2bf(av.x, av.y), pk2bf(av.z, av.w));
      float4 bv = *(const float4*)(W + (size_t)row * E_ + kk + c4);
      *(uint2*)(&Bsm[row * 40 + c4]) = make_uint2(pk2bf(bv.x, bv.y), pk2bf(bv.z, bv.w));
    }
    __syncthreads();

    FragBF a[2];
    #pragma unroll
    for (int mt = 0; mt < 2; ++mt) {
      // A 16x32 layout: lanes0-15 K in {0..7,16..23}, lanes16-31 K in {8..15,24..31}
      const bf16_t* ar = &Asm[(wm * 32 + mt * 16 + ln) * 40 + (half ? 8 : 0)];
      a[mt].q[0] = *(const uint4*)(ar);
      a[mt].q[1] = *(const uint4*)(ar + 16);
    }
    #pragma unroll
    for (int nt = 0; nt < 4; ++nt) {
      FragBF bfr;
      // B 32x16 layout: lane = N, lanes0-15 K in {0..7,16..23}, lanes16-31 K in {8..15,24..31}
      const bf16_t* br = &Bsm[(wn * 64 + nt * 16 + ln) * 40 + (half ? 8 : 0)];
      bfr.q[0] = *(const uint4*)(br);
      bfr.q[1] = *(const uint4*)(br + 16);
      #pragma unroll
      for (int mt = 0; mt < 2; ++mt)
        acc[mt][nt] = __builtin_amdgcn_wmma_f32_16x16x32_bf16(
            false, a[mt].v, false, bfr.v, (short)0, acc[mt][nt], false, false);
    }
  }

  // epilogue: scatter to bf16 [B,H,S,D] (q,k) or [B,H,D,S] (v)
  #pragma unroll
  for (int nt = 0; nt < 4; ++nt) {
    int eg   = n0g + wn * 64 + nt * 16 + ln;  // 0..3071
    int e    = eg & 1023;
    float bi = b_in[eg];
    int h = e >> 6, d = e & 63;
    #pragma unroll
    for (int mt = 0; mt < 2; ++mt) {
      #pragma unroll
      for (int r = 0; r < 8; ++r) {
        int m  = m0 + wm * 32 + mt * 16 + r + 8 * half;  // C layout: m = r + 8*half
        int bb = m >> 10, s = m & 1023;
        size_t idx = (t == 2)
            ? ((((size_t)(bb * H_ + h)) * D_ + d) * S_ + s)    // V: [B,H,D,S]
            : ((((size_t)(bb * H_ + h)) * S_ + s) * D_ + d);   // Q,K: [B,H,S,D]
        dst[idx] = f2bf(acc[mt][nt][r] + bi);
      }
    }
  }
}

// ---------------------------------------------------------------------------
// Kernel 2: flash attention per (b,h).  Block = 128 q-rows, 8 waves x 16 rows.
// Double-buffered K/V tiles loaded with async global->LDS; V already [D,S] so
// B-fragments read contiguous keys; online softmax; P staged through LDS.
// ---------------------------------------------------------------------------
__global__ __launch_bounds__(256) void mha_attn(
    const bf16_t* __restrict__ qh, const bf16_t* __restrict__ kh,
    const bf16_t* __restrict__ vt, bf16_t* __restrict__ ctx)
{
  __shared__ __align__(16) bf16_t Ksm[2][64 * 72];    // [key][d], row 144B
  __shared__ __align__(16) bf16_t Vsm[2][64 * 72];    // [d][key]
  __shared__ __align__(16) bf16_t Psm[8 * 16 * 72];   // per-wave P staging

  const int hb  = blockIdx.x;                 // b*H + h
  const int q0  = blockIdx.y * 128;
  const int tid = threadIdx.x;
  const int w = tid >> 5, lane = tid & 31, ln = lane & 15, half = lane >> 4;

  const bf16_t* __restrict__ qhead  = qh + (size_t)hb * S_ * D_;
  const bf16_t* __restrict__ khead  = kh + (size_t)hb * S_ * D_;
  const bf16_t* __restrict__ vthead = vt + (size_t)hb * D_ * S_;

  // Q fragments for this wave's 16 rows (D=64 -> two K=32 steps), kept in VGPRs
  FragBF aq[2];
  {
    const bf16_t* qrow = qhead + (size_t)(q0 + w * 16 + ln) * D_;
    const int c0 = half ? 8 : 0;
    #pragma unroll
    for (int j = 0; j < 2; ++j) {
      aq[j].q[0] = *(const uint4*)(qrow + 32 * j + c0);
      aq[j].q[1] = *(const uint4*)(qrow + 32 * j + 16 + c0);
    }
  }

  auto issue_tiles = [&](int kt, int buf) {
    #pragma unroll
    for (int i = 0; i < 2; ++i) {
      int idx = i * 256 + tid;        // 512 uint4 slots per tile
      int row = idx >> 3;             // 0..63
      int col = (idx & 7) * 8;        // 0..56
      // K tile: [key][d]
      async_copy16(khead + (size_t)(kt * 64 + row) * D_ + col,
                   &Ksm[buf][row * 72 + col]);
      // V tile: [d][key] (vt is [D,S], rows contiguous over keys)
      async_copy16(vthead + (size_t)row * S_ + kt * 64 + col,
                   &Vsm[buf][row * 72 + col]);
    }
  };

  v8f o[4] = {};
  float m_run[8], l_run[8];
  #pragma unroll
  for (int r = 0; r < 8; ++r) { m_run[r] = -3.0e38f; l_run[r] = 0.f; }

  const float scale = 0.125f;  // D^-0.5
  bf16_t* pw = &Psm[w * 16 * 72];

  issue_tiles(0, 0);
  wait_async_lds();
  __syncthreads();

  constexpr int NT = S_ / 64;
  for (int kt = 0; kt < NT; ++kt) {
    const int cur = kt & 1;
    if (kt + 1 < NT) issue_tiles(kt + 1, cur ^ 1);   // pipeline next tile
    if (kt + 2 < NT) {                                // L2 prefetch tile kt+2
      __builtin_prefetch(khead + (size_t)((kt + 2) * 64 + (tid & 63)) * D_, 0, 0);
      __builtin_prefetch(vthead + (size_t)(tid & 63) * S_ + (kt + 2) * 64, 0, 0);
    }
    const bf16_t* Kb = Ksm[cur];
    const bf16_t* Vb = Vsm[cur];

    // scores: S(16x64) = Q(16x64) . K^T, 8 WMMA
    v8f sc[4] = {};
    #pragma unroll
    for (int j = 0; j < 2; ++j) {
      #pragma unroll
      for (int nt = 0; nt < 4; ++nt) {
        FragBF bk;
        const bf16_t* kr = &Kb[(nt * 16 + ln) * 72 + 32 * j + (half ? 8 : 0)];
        bk.q[0] = *(const uint4*)(kr);
        bk.q[1] = *(const uint4*)(kr + 16);
        sc[nt] = __builtin_amdgcn_wmma_f32_16x16x32_bf16(
            false, aq[j].v, false, bk.v, (short)0, sc[nt], false, false);
      }
    }

    // online softmax; C layout: row m = r + 8*half lives across lanes (lane%16 = n)
    #pragma unroll
    for (int r = 0; r < 8; ++r) {
      float mx = fmaxf(fmaxf(sc[0][r], sc[1][r]), fmaxf(sc[2][r], sc[3][r]));
      #pragma unroll
      for (int off = 1; off < 16; off <<= 1)
        mx = fmaxf(mx, __shfl_xor(mx, off, 16));
      float mnew  = fmaxf(m_run[r], mx * scale);
      float alpha = __expf(m_run[r] - mnew);
      m_run[r] = mnew;
      float p0 = __expf(sc[0][r] * scale - mnew);
      float p1 = __expf(sc[1][r] * scale - mnew);
      float p2 = __expf(sc[2][r] * scale - mnew);
      float p3 = __expf(sc[3][r] * scale - mnew);
      sc[0][r] = p0; sc[1][r] = p1; sc[2][r] = p2; sc[3][r] = p3;
      float ps = (p0 + p1) + (p2 + p3);
      #pragma unroll
      for (int off = 1; off < 16; off <<= 1)
        ps += __shfl_xor(ps, off, 16);
      l_run[r] = l_run[r] * alpha + ps;
      #pragma unroll
      for (int nt = 0; nt < 4; ++nt) o[nt][r] *= alpha;
    }

    // stage P (C layout) -> LDS -> A-fragment layout
    #pragma unroll
    for (int nt = 0; nt < 4; ++nt)
      #pragma unroll
      for (int r = 0; r < 8; ++r)
        pw[(r + 8 * half) * 72 + nt * 16 + ln] = f2bf(sc[nt][r]);
    __syncthreads();

    // O += P(16x64) . V(64x64), 8 WMMA
    #pragma unroll
    for (int j = 0; j < 2; ++j) {
      FragBF ap;
      const bf16_t* pr = pw + ln * 72 + 32 * j + (half ? 8 : 0);
      ap.q[0] = *(const uint4*)(pr);
      ap.q[1] = *(const uint4*)(pr + 16);
      #pragma unroll
      for (int nt = 0; nt < 4; ++nt) {
        FragBF bv2;
        const bf16_t* vr = &Vb[(nt * 16 + ln) * 72 + 32 * j + (half ? 8 : 0)];
        bv2.q[0] = *(const uint4*)(vr);
        bv2.q[1] = *(const uint4*)(vr + 16);
        o[nt] = __builtin_amdgcn_wmma_f32_16x16x32_bf16(
            false, ap.v, false, bv2.v, (short)0, o[nt], false, false);
      }
    }

    wait_async_lds();   // next tile's async copies landed (this wave's share)
    __syncthreads();    // everyone done reading cur + done writing nxt
  }

  // epilogue: normalize, write ctx as bf16 [B,S,E] with e = h*64 + d
  const int b = hb >> 4, h = hb & 15;
  #pragma unroll
  for (int nt = 0; nt < 4; ++nt) {
    #pragma unroll
    for (int r = 0; r < 8; ++r) {
      int s = q0 + w * 16 + r + 8 * half;
      float val = o[nt][r] / l_run[r];
      ctx[((size_t)(b * S_ + s)) * E_ + h * 64 + nt * 16 + ln] = f2bf(val);
    }
  }
}

// ---------------------------------------------------------------------------
// Kernel 3: out-projection.  out[m,n] = ctx[m,:] . W_out[n,:] + bias[n]  (fp32)
// A tile (bf16 ctx) copied with async global->LDS, overlapped with the fp32->
// bf16 conversion of the weight tile.
// ---------------------------------------------------------------------------
__global__ __launch_bounds__(256) void mha_outproj(
    const bf16_t* __restrict__ ctx, const float* __restrict__ W,
    const float* __restrict__ bias, float* __restrict__ out)
{
  __shared__ __align__(16) bf16_t Asm[128 * 40];
  __shared__ __align__(16) bf16_t Bsm[128 * 40];

  const int m0 = blockIdx.x * 128;
  const int n0 = blockIdx.y * 128;
  const int tid  = threadIdx.x;
  const int w    = tid >> 5;
  const int lane = tid & 31;
  const int ln   = lane & 15;
  const int half = lane >> 4;
  const int wm   = w >> 1;
  const int wn   = w & 1;

  v8f acc[2][4] = {};

  for (int kk = 0; kk < E_; kk += 32) {
    __syncthreads();
    #pragma unroll
    for (int i = 0; i < 2; ++i) {       // A tile: bf16 straight async copy
      int li  = i * 256 + tid;          // 512 uint4 slots
      int row = li >> 2;                // 0..127
      int c8  = (li & 3) * 8;           // 0,8,16,24
      async_copy16(ctx + (size_t)(m0 + row) * E_ + kk + c8, &Asm[row * 40 + c8]);
    }
    #pragma unroll
    for (int i = 0; i < 4; ++i) {       // B tile: fp32 -> bf16 (overlaps async)
      int li  = i * 256 + tid;
      int row = li >> 3;
      int c4  = (li & 7) * 4;
      float4 bv = *(const float4*)(W + (size_t)(n0 + row) * E_ + kk + c4);
      *(uint2*)(&Bsm[row * 40 + c4]) = make_uint2(pk2bf(bv.x, bv.y), pk2bf(bv.z, bv.w));
    }
    wait_async_lds();
    __syncthreads();

    FragBF a[2];
    #pragma unroll
    for (int mt = 0; mt < 2; ++mt) {
      const bf16_t* ar = &Asm[(wm * 32 + mt * 16 + ln) * 40 + (half ? 8 : 0)];
      a[mt].q[0] = *(const uint4*)(ar);
      a[mt].q[1] = *(const uint4*)(ar + 16);
    }
    #pragma unroll
    for (int nt = 0; nt < 4; ++nt) {
      FragBF bfr;
      const bf16_t* br = &Bsm[(wn * 64 + nt * 16 + ln) * 40 + (half ? 8 : 0)];
      bfr.q[0] = *(const uint4*)(br);
      bfr.q[1] = *(const uint4*)(br + 16);
      #pragma unroll
      for (int mt = 0; mt < 2; ++mt)
        acc[mt][nt] = __builtin_amdgcn_wmma_f32_16x16x32_bf16(
            false, a[mt].v, false, bfr.v, (short)0, acc[mt][nt], false, false);
    }
  }

  #pragma unroll
  for (int nt = 0; nt < 4; ++nt) {
    int n    = n0 + wn * 64 + nt * 16 + ln;
    float bi = bias[n];
    #pragma unroll
    for (int mt = 0; mt < 2; ++mt) {
      #pragma unroll
      for (int r = 0; r < 8; ++r) {
        int m = m0 + wm * 32 + mt * 16 + r + 8 * half;
        out[(size_t)m * E_ + n] = acc[mt][nt][r] + bi;
      }
    }
  }
}

// ---------------------------------------------------------------------------
extern "C" void kernel_launch(void* const* d_in, const int* in_sizes, int n_in,
                              void* d_out, int out_size, void* d_ws, size_t ws_size,
                              hipStream_t stream) {
  (void)in_sizes; (void)n_in; (void)out_size; (void)ws_size;
  const float* q     = (const float*)d_in[0];
  const float* k     = (const float*)d_in[1];
  const float* v     = (const float*)d_in[2];
  const float* w_in  = (const float*)d_in[3];
  const float* b_in  = (const float*)d_in[4];
  const float* w_out = (const float*)d_in[5];
  const float* b_out = (const float*)d_in[6];
  float* out = (float*)d_out;

  const size_t NE = (size_t)B_ * H_ * S_ * D_;  // 8388608 elements
  bf16_t* qh  = (bf16_t*)d_ws;                  // [B,H,S,D] bf16
  bf16_t* kh  = qh + NE;                        // [B,H,S,D] bf16
  bf16_t* vt  = kh + NE;                        // [B,H,D,S] bf16 (pre-transposed)
  bf16_t* ctx = vt + NE;                        // [B,S,E]  bf16; 64 MB ws total

  // 1) packed in-projection: 8192x3072 output, tiles 128x128
  mha_inproj<<<dim3(64, 24), 256, 0, stream>>>(q, k, v, w_in, b_in, qh, kh, vt);
  // 2) flash attention: 128 heads x 8 q-tiles of 128 rows
  mha_attn<<<dim3(128, 8), 256, 0, stream>>>(qh, kh, vt, ctx);
  // 3) out-projection: 8192x1024 fp32 output
  mha_outproj<<<dim3(64, 8), 256, 0, stream>>>(ctx, w_out, b_out, out);
}
